// HelioAttention_2422361555506
// MI455X (gfx1250) — compile-verified, matrix operand order checked by reading, not run on hardware
//
#include <hip/hip_runtime.h>
#include <hip/hip_bf16.h>
#include <math.h>

// ---------------------------------------------------------------------------
// HelioAttention forward on MI455X (gfx1250, wave32, WMMA)
//   B=2, S=2048, D=1024, H=16, hd=64
// v_wmma_f32_16x16x32_f16 everywhere; GEMMs double-buffer tiles in LDS via
// global_load_async_to_lds_b128 (ASYNCcnt); softmax row reductions via DPP.
// Flash attention processes 2 query tiles per wave to double WMMA/byte.
// ---------------------------------------------------------------------------

typedef _Float16 h16;
typedef __attribute__((ext_vector_type(16))) _Float16 v16h;
typedef __attribute__((ext_vector_type(8)))  _Float16 v8h;
typedef __attribute__((ext_vector_type(4)))  _Float16 v4h;
typedef __attribute__((ext_vector_type(8)))  float    v8f;

#define WMMA16x16x32(a, b, c) \
  __builtin_amdgcn_wmma_f32_16x16x32_f16(false, (a), false, (b), (short)0, (c), false, false)

constexpr int Bz = 2;
constexpr int Sq = 2048;
constexpr int Dm = 1024;
constexpr int Hh = 16;
constexpr int HD = 64;
constexpr int Mrows = Bz * Sq;   // 4096
constexpr int LDT = 40;          // padded LDS tile row stride (halfs): 20*r mod 64 distinct

// 16-lane (half-wave) reductions via DPP: xor1, xor2, half-mirror, mirror.
// Rows of the WMMA C-matrix live in 16-lane halves, so this reduces each row.
__device__ __forceinline__ float dpp_hmax16(float x) {
  float t;
  t = __int_as_float(__builtin_amdgcn_update_dpp(__float_as_int(x), __float_as_int(x), 0x0B1, 0xF, 0xF, true)); x = fmaxf(x, t);
  t = __int_as_float(__builtin_amdgcn_update_dpp(__float_as_int(x), __float_as_int(x), 0x04E, 0xF, 0xF, true)); x = fmaxf(x, t);
  t = __int_as_float(__builtin_amdgcn_update_dpp(__float_as_int(x), __float_as_int(x), 0x141, 0xF, 0xF, true)); x = fmaxf(x, t);
  t = __int_as_float(__builtin_amdgcn_update_dpp(__float_as_int(x), __float_as_int(x), 0x140, 0xF, 0xF, true)); x = fmaxf(x, t);
  return x;
}
__device__ __forceinline__ float dpp_hsum16(float x) {
  float t;
  t = __int_as_float(__builtin_amdgcn_update_dpp(__float_as_int(x), __float_as_int(x), 0x0B1, 0xF, 0xF, true)); x += t;
  t = __int_as_float(__builtin_amdgcn_update_dpp(__float_as_int(x), __float_as_int(x), 0x04E, 0xF, 0xF, true)); x += t;
  t = __int_as_float(__builtin_amdgcn_update_dpp(__float_as_int(x), __float_as_int(x), 0x141, 0xF, 0xF, true)); x += t;
  t = __int_as_float(__builtin_amdgcn_update_dpp(__float_as_int(x), __float_as_int(x), 0x140, 0xF, 0xF, true)); x += t;
  return x;
}

// Async 16-byte global -> LDS copy (ASYNCcnt), per-lane addresses.
__device__ __forceinline__ void async_g2l_b128(unsigned lds_off, const void* gptr) {
  unsigned long long ga = (unsigned long long)(uintptr_t)gptr;
  asm volatile("global_load_async_to_lds_b128 %0, %1, off" :: "v"(lds_off), "v"(ga) : "memory");
}
__device__ __forceinline__ unsigned lds_offset(const void* p) {
  return (unsigned)(uintptr_t)p;   // flat LDS addr truncates to LDS offset
}

// ---------------------------------------------------------------------------
// Load a 16(rows) x 32(k) f16 tile from row-major memory into WMMA A-operand
// layout (ISA 7.12.2, 16-bit A 16x32):
//   lanes 0-15  : row = lane,     K = 0-7   and 16-23
//   lanes 16-31 : row = lane-16,  K = 8-15  and 24-31
// The B operand of D = A*B is the A-layout of B^T, so K tiles / W rows /
// V^T rows all load through this same helper.
// ---------------------------------------------------------------------------
__device__ __forceinline__ v16h load_af(const h16* __restrict__ p, int ld, int k0, int lane) {
  const int row = lane & 15;
  const int hi  = (lane >> 4) << 3;           // 0 or 8
  const h16* q = p + (size_t)row * ld + k0 + hi;
  union { v16h v; v8h h[2]; } u;
  u.h[0] = *(const v8h*)(q);
  u.h[1] = *(const v8h*)(q + 16);
  return u.v;
}

// ---------------------------------------------------------------------------
// f32 -> f16 conversion (X and the four weight matrices), 4 elems/thread
// ---------------------------------------------------------------------------
__global__ __launch_bounds__(256) void cvt_f32_f16(const float* __restrict__ src,
                                                   h16* __restrict__ dst, int n) {
  int i = (blockIdx.x * blockDim.x + threadIdx.x) * 4;
  if (i + 3 < n) {
    float4 v = *(const float4*)(src + i);
    v4h o = { (h16)v.x, (h16)v.y, (h16)v.z, (h16)v.w };
    *(v4h*)(dst + i) = o;
  }
}

// ---------------------------------------------------------------------------
// GEMM: Y = A(4096x1024, f16 row-major) * W^T (W is 1024x1024 f16 row-major).
// Block = 128x128 (8 waves as 4Mx2N), wave = 32x64 -> 2x4 accum tiles.
// Operand tiles double-buffered in LDS with async b128 copies:
//   wait(cur) -> barrier -> issue(next into alt buffer) -> compute(cur)
// MODE 0: store f16 head-major [B,H,S,64]      (Q, K)
// MODE 1: store f16 head-transposed [B,H,64,S] (V^T, feeds PV B-operand)
// MODE 2: store f32 row-major [4096,1024]      (final output projection)
// ---------------------------------------------------------------------------
template <int MODE>
__global__ __launch_bounds__(256) void gemm_xwt(const h16* __restrict__ A,
                                                const h16* __restrict__ W,
                                                void* __restrict__ dst) {
  __shared__ h16 shA[2][128 * LDT];   // 2 x 10 KiB
  __shared__ h16 shW[2][128 * LDT];   // 2 x 10 KiB

  const int tid  = threadIdx.x;
  const int lane = tid & 31;
  const int w    = tid >> 5;
  const int wm   = w & 3;            // 0..3
  const int wn   = w >> 2;           // 0..1
  const int rowB = blockIdx.y * 128;           // block row origin
  const int colB = blockIdx.x * 128;           // block col origin (W row origin)

  // async-copy assignment: each thread moves 2x16B of A and 2x16B of W
  const int crow  = tid >> 1;                  // 0..127
  const int csub0 = (tid & 1) * 2;             // subs {0,1} or {2,3} (16B units)
  const h16* agrow = A + (size_t)(rowB + crow) * Dm;
  const h16* wgrow = W + (size_t)(colB + crow) * Dm;
  const unsigned laA0 = lds_offset(&shA[0][crow * LDT + csub0 * 8]);
  const unsigned laW0 = lds_offset(&shW[0][crow * LDT + csub0 * 8]);
  const unsigned bufstride = (unsigned)(128 * LDT * sizeof(h16));

  auto issue_tile = [&](int k0, int buf) {
    const unsigned bo = (unsigned)buf * bufstride;
#pragma unroll
    for (int c = 0; c < 2; ++c) {
      const int sub = csub0 + c;
      async_g2l_b128(laA0 + bo + c * 16, agrow + k0 + sub * 8);
      async_g2l_b128(laW0 + bo + c * 16, wgrow + k0 + sub * 8);
    }
  };

  v8f acc[2][4];
#pragma unroll
  for (int i = 0; i < 2; ++i)
#pragma unroll
    for (int j = 0; j < 4; ++j) acc[i][j] = {};

  issue_tile(0, 0);                            // prologue fill

  for (int kt = 0; kt < Dm / 32; ++kt) {
    const int cur = kt & 1;
    asm volatile("s_wait_asynccnt 0x0" ::: "memory");
    __syncthreads();                           // cur tile visible to all waves
    if (kt + 1 < Dm / 32) issue_tile((kt + 1) * 32, cur ^ 1);  // overlap copy

    v16h af[2], bf[4];
#pragma unroll
    for (int i = 0; i < 2; ++i)
      af[i] = load_af(shA[cur] + (wm * 32 + 16 * i) * LDT, LDT, 0, lane);
#pragma unroll
    for (int j = 0; j < 4; ++j)
      bf[j] = load_af(shW[cur] + (wn * 64 + 16 * j) * LDT, LDT, 0, lane);

#pragma unroll
    for (int i = 0; i < 2; ++i)
#pragma unroll
      for (int j = 0; j < 4; ++j)
        acc[i][j] = WMMA16x16x32(af[i], bf[j], acc[i][j]);
  }

  // C/D layout: lane<16 -> N=lane, M=r ; lane>=16 -> N=lane-16, M=r+8
  const int nlo   = lane & 15;
  const int rbase = (lane >> 4) * 8;
  const int row0  = rowB + wm * 32;
  const int col0  = colB + wn * 64;
#pragma unroll
  for (int i = 0; i < 2; ++i) {
#pragma unroll
    for (int j = 0; j < 4; ++j) {
#pragma unroll
      for (int r = 0; r < 8; ++r) {
        const int grow = row0 + 16 * i + rbase + r;     // 0..4095
        const int gcol = col0 + 16 * j + nlo;           // 0..1023
        const float val = acc[i][j][r];
        if constexpr (MODE == 2) {
          ((float*)dst)[(size_t)grow * Dm + gcol] = val;
        } else {
          const int b = grow >> 11, s = grow & (Sq - 1);
          const int h = gcol >> 6, d = gcol & (HD - 1);
          if constexpr (MODE == 0)
            ((h16*)dst)[(((size_t)b * Hh + h) * Sq + s) * HD + d] = (h16)val;
          else
            ((h16*)dst)[(((size_t)b * Hh + h) * HD + d) * Sq + s] = (h16)val;
        }
      }
    }
  }
}

// ---------------------------------------------------------------------------
// Flash attention: 1 wave per (b, h, 32-query block) = 2048 waves.
// Each wave owns TWO 16-row query tiles so every K/V fragment load feeds
// 2x the WMMAs. 32 keys / iteration: 8 score WMMAs, DPP row-max softmax
// (lane-local running row-sums, reduced once at the end), P staged through
// per-wave LDS (C-layout -> A-layout), 8 PV WMMAs sharing the V fragments.
// ---------------------------------------------------------------------------
__global__ __launch_bounds__(256) void attn_fwd(const h16* __restrict__ Qh,
                                                const h16* __restrict__ Kh,
                                                const h16* __restrict__ Vt,
                                                const int* __restrict__ amask,
                                                h16* __restrict__ Ah) {
  __shared__ h16 lds_p[8][2][16][32];    // per-wave 2x(16x32) P staging (16 KiB)

  const int lane = threadIdx.x & 31;
  const int w    = threadIdx.x >> 5;
  const int gw   = blockIdx.x * 8 + w;          // 0..2047
  const int b    = gw >> 10;                    // / (H*64)
  const int rem  = gw & 1023;
  const int h    = rem >> 6;
  const int q0   = (rem & 63) * 32;

  const h16* qbase = Qh + (((size_t)b * Hh + h) * Sq + q0) * HD;
  const h16* kbase = Kh + (((size_t)b * Hh + h) * Sq) * HD;
  const h16* vbase = Vt + (((size_t)b * Hh + h) * HD) * Sq;
  const int* mrow  = amask + b * Sq;

  v16h qf[2][2];
#pragma unroll
  for (int u = 0; u < 2; ++u) {
    qf[u][0] = load_af(qbase + (size_t)u * 16 * HD, HD, 0, lane);
    qf[u][1] = load_af(qbase + (size_t)u * 16 * HD, HD, 32, lane);
  }

  float rm[2][8], rl[2][8];              // rl is LANE-LOCAL partial row sum
  v8f o[2][4];
#pragma unroll
  for (int u = 0; u < 2; ++u) {
#pragma unroll
    for (int r = 0; r < 8; ++r) { rm[u][r] = -INFINITY; rl[u][r] = 0.0f; }
#pragma unroll
    for (int j = 0; j < 4; ++j) o[u][j] = {};
  }

  const int csub = lane & 15;
  const int rh   = (lane >> 4) * 8;

  for (int kt = 0; kt < Sq / 32; ++kt) {
    const int k0 = kt * 32;

    // ---- shared K fragments for both query tiles ----
    v16h kb[2][2];
#pragma unroll
    for (int t = 0; t < 2; ++t) {
      kb[t][0] = load_af(kbase + (size_t)(k0 + 16 * t) * HD, HD, 0, lane);
      kb[t][1] = load_af(kbase + (size_t)(k0 + 16 * t) * HD, HD, 32, lane);
    }
    const int mv0 = mrow[k0 + csub];
    const int mv1 = mrow[k0 + 16 + csub];

#pragma unroll
    for (int u = 0; u < 2; ++u) {
      // ---- scores (4 WMMAs per query tile) ----
      v8f sc[2];
#pragma unroll
      for (int t = 0; t < 2; ++t) {
        sc[t] = {};
        sc[t] = WMMA16x16x32(qf[u][0], kb[t][0], sc[t]);
        sc[t] = WMMA16x16x32(qf[u][1], kb[t][1], sc[t]);
      }
      // ---- online softmax: DPP row-max, lane-local row-sum ----
#pragma unroll
      for (int r = 0; r < 8; ++r) {
        float s0 = mv0 ? sc[0][r] * 0.125f : -1e9f;   // 1/sqrt(64)
        float s1 = mv1 ? sc[1][r] * 0.125f : -1e9f;
        const float m   = dpp_hmax16(fmaxf(s0, s1));  // uniform across half
        const float mn  = fmaxf(rm[u][r], m);
        const float cor = __expf(rm[u][r] - mn);
        const float p0  = __expf(s0 - mn);
        const float p1  = __expf(s1 - mn);
        rl[u][r] = rl[u][r] * cor + (p0 + p1);        // reduce at end
        rm[u][r] = mn;
        o[u][0][r] *= cor; o[u][1][r] *= cor; o[u][2][r] *= cor; o[u][3][r] *= cor;
        lds_p[w][u][rh + r][csub]      = (h16)p0;
        lds_p[w][u][rh + r][16 + csub] = (h16)p1;
      }
    }
    asm volatile("s_wait_dscnt 0x0" ::: "memory");

    // ---- reload P as 16x32 A-operands ----
    v16h pf[2];
    {
      const int prow = lane & 15;
      const int phi  = (lane >> 4) << 3;
#pragma unroll
      for (int u = 0; u < 2; ++u) {
        union { v16h v; v8h hh[2]; } uu;
        uu.hh[0] = *(const v8h*)&lds_p[w][u][prow][phi];
        uu.hh[1] = *(const v8h*)&lds_p[w][u][prow][16 + phi];
        pf[u] = uu.v;
      }
    }
    // ---- P*V: V fragments shared by both query tiles (8 WMMAs) ----
#pragma unroll
    for (int j = 0; j < 4; ++j) {
      v16h vb = load_af(vbase + (size_t)(j * 16) * Sq, Sq, k0, lane);
      o[0][j] = WMMA16x16x32(pf[0], vb, o[0][j]);
      o[1][j] = WMMA16x16x32(pf[1], vb, o[1][j]);
    }
  }

  // ---- final row-sum reduction, normalize, store f16 Ah[4096,1024] ----
#pragma unroll
  for (int u = 0; u < 2; ++u) {
#pragma unroll
    for (int r = 0; r < 8; ++r) {
      const float inv = 1.0f / dpp_hsum16(rl[u][r]);
      const int s = q0 + u * 16 + rh + r;
#pragma unroll
      for (int j = 0; j < 4; ++j) {
        const int col = h * HD + j * 16 + csub;
        Ah[((size_t)b * Sq + s) * Dm + col] = (h16)(o[u][j][r] * inv);
      }
    }
  }
}

// ---------------------------------------------------------------------------
// Launch
// ---------------------------------------------------------------------------
extern "C" void kernel_launch(void* const* d_in, const int* in_sizes, int n_in,
                              void* d_out, int out_size, void* d_ws, size_t ws_size,
                              hipStream_t stream) {
  const float* X    = (const float*)d_in[0];
  const int*   mask = (const int*)d_in[1];
  const float* Wq   = (const float*)d_in[2];
  const float* Wk   = (const float*)d_in[3];
  const float* Wv   = (const float*)d_in[4];
  const float* Wo   = (const float*)d_in[5];

  char* ws = (char*)d_ws;
  const size_t MiB = 1u << 20;
  h16* Xh  = (h16*)(ws + 0 * MiB);   // 8 MiB  [4096,1024]
  h16* Wqh = (h16*)(ws + 8 * MiB);   // 2 MiB
  h16* Wkh = (h16*)(ws + 10 * MiB);  // 2 MiB
  h16* Wvh = (h16*)(ws + 12 * MiB);  // 2 MiB
  h16* Woh = (h16*)(ws + 14 * MiB);  // 2 MiB
  h16* Qh  = (h16*)(ws + 16 * MiB);  // 8 MiB  [B,H,S,64]
  h16* Kh  = (h16*)(ws + 24 * MiB);  // 8 MiB  [B,H,S,64]
  h16* Vt  = (h16*)(ws + 32 * MiB);  // 8 MiB  [B,H,64,S]
  h16* Ah  = (h16*)(ws + 40 * MiB);  // 8 MiB  [4096,1024]

  // f32 -> f16
  cvt_f32_f16<<<(Mrows * Dm) / 1024, 256, 0, stream>>>(X, Xh, Mrows * Dm);
  cvt_f32_f16<<<(Dm * Dm) / 1024, 256, 0, stream>>>(Wq, Wqh, Dm * Dm);
  cvt_f32_f16<<<(Dm * Dm) / 1024, 256, 0, stream>>>(Wk, Wkh, Dm * Dm);
  cvt_f32_f16<<<(Dm * Dm) / 1024, 256, 0, stream>>>(Wv, Wvh, Dm * Dm);
  cvt_f32_f16<<<(Dm * Dm) / 1024, 256, 0, stream>>>(Wo, Woh, Dm * Dm);

  // QKV projections
  dim3 gg(Dm / 128, Mrows / 128);    // (8, 32)
  gemm_xwt<0><<<gg, 256, 0, stream>>>(Xh, Wqh, Qh);
  gemm_xwt<0><<<gg, 256, 0, stream>>>(Xh, Wkh, Kh);
  gemm_xwt<1><<<gg, 256, 0, stream>>>(Xh, Wvh, Vt);

  // streaming attention (2048 waves, 8 waves/block)
  attn_fwd<<<256, 256, 0, stream>>>(Qh, Kh, Vt, mask, Ah);

  // output projection -> f32 d_out
  gemm_xwt<2><<<gg, 256, 0, stream>>>(Ah, Woh, d_out);
}
